// TorchMmoeModel_23562190586007
// MI455X (gfx1250) — matrix operand order repaired and checked by reading, not compiled, not run on hardware
//
#include <hip/hip_runtime.h>

typedef __attribute__((ext_vector_type(16))) __bf16 v16bf;
typedef __attribute__((ext_vector_type(8)))  __bf16 v8bf;
typedef __attribute__((ext_vector_type(4)))  __bf16 v4bf;
typedef __attribute__((ext_vector_type(8)))  float  v8f;
typedef __attribute__((ext_vector_type(4)))  float  v4f;
typedef __attribute__((ext_vector_type(4)))  int    v4i;

// Problem sizes (from reference setup_inputs)
static constexpr int  kB  = 32768;
static constexpr int  kD  = 1472;
static constexpr int  kE  = 8;
static constexpr int  kH1 = 512;
static constexpr int  kH2 = 256;
static constexpr int  kT  = 2;
static constexpr float kEps = 1e-5f;

// ---------------------------------------------------------------------------
// CDNA5 async global->LDS copy (bypasses VGPRs, tracked by ASYNCcnt).
// Falls back to a synchronous reg round-trip if the builtin is unavailable.
// Builtin prototype (from compiler diagnostic): param0 = v4i in addrspace(1),
// param1 = v4i in addrspace(3), then imm offset, imm cpol.
// ---------------------------------------------------------------------------
#if defined(__HIP_DEVICE_COMPILE__) && __has_builtin(__builtin_amdgcn_global_load_async_to_lds_b128)
#define HAS_ASYNC_LDS 1
#else
#define HAS_ASYNC_LDS 0
#endif

#if HAS_ASYNC_LDS
typedef __attribute__((address_space(1))) v4i* gbl_v4i_p;
typedef __attribute__((address_space(3))) v4i* lds_v4i_p;
#endif

__device__ __forceinline__ void copy16_to_lds(const __bf16* g, __bf16* l) {
#if HAS_ASYNC_LDS
    __builtin_amdgcn_global_load_async_to_lds_b128(
        (gbl_v4i_p)(uintptr_t)g, (lds_v4i_p)(uintptr_t)l, 0, 0);
#else
    *(v8bf*)l = *(const v8bf*)g;
#endif
}

__device__ __forceinline__ void async_wait_all() {
#if HAS_ASYNC_LDS
#if __has_builtin(__builtin_amdgcn_s_wait_asynccnt)
    __builtin_amdgcn_s_wait_asynccnt(0);
#else
    asm volatile("s_wait_asynccnt 0x0" ::: "memory");
#endif
#endif
}

// ---------------------------------------------------------------------------
// small utility kernels
// ---------------------------------------------------------------------------
__global__ void zero_f32_kernel(float* __restrict__ p, int n) {
    int i = blockIdx.x * blockDim.x + threadIdx.x;
    if (i < n) p[i] = 0.f;
}

__global__ void cast_f32_to_bf16_kernel(const float* __restrict__ in,
                                        __bf16* __restrict__ out, long n4) {
    long i = (long)blockIdx.x * blockDim.x + threadIdx.x;
    if (i >= n4) return;
    v4f v = *(const v4f*)(in + i * 4);
    v4bf o;
#pragma unroll
    for (int j = 0; j < 4; ++j) o[j] = (__bf16)v[j];
    *(v4bf*)(out + i * 4) = o;
}

// in: [E][R][C] fp32  ->  out: [E][C][R] bf16  (out flat index has R innermost)
__global__ void transpose_cast_kernel(const float* __restrict__ in,
                                      __bf16* __restrict__ out,
                                      int R, int C, long total) {
    long i = (long)blockIdx.x * blockDim.x + threadIdx.x;
    if (i >= total) return;
    int  d  = (int)(i % R);
    long t2 = i / R;
    int  h  = (int)(t2 % C);
    int  e  = (int)(t2 / C);
    out[i] = (__bf16)in[((long)e * R + d) * C + h];
}

__global__ void bn_finalize_kernel(const float* __restrict__ sum,
                                   const float* __restrict__ sumsq,
                                   const float* __restrict__ gamma,
                                   const float* __restrict__ beta,
                                   float* __restrict__ aOut,
                                   float* __restrict__ bOut,
                                   int count, float invB) {
    int i = blockIdx.x * blockDim.x + threadIdx.x;
    if (i >= count) return;
    float mean = sum[i] * invB;
    float var  = fmaxf(sumsq[i] * invB - mean * mean, 0.f);
    float a    = gamma[i] * rsqrtf(var + kEps);
    aOut[i] = a;
    bOut[i] = beta[i] - a * mean;
}

// ---------------------------------------------------------------------------
// WMMA bf16 GEMM: C[e][m][n] = A[e][m][:] . Bt[e][n][:] + bias[e][n]
// Optionally applies relu(aScale*x + aShift) to A elements while staging
// (BN+ReLU of the previous layer, scale indexed along K).
// Accumulates per-(e,n) sum / sum-of-squares for the following BatchNorm.
// Block tile 128x128, BK=32, 256 threads (8 waves), wave tile 32x64.
// Double-buffered LDS; next tile streamed via GLOBAL_LOAD_ASYNC_TO_LDS_B128.
// ---------------------------------------------------------------------------
__global__ __launch_bounds__(256)
void gemm_bn_stats_kernel(const __bf16* __restrict__ A,  long aBatch,
                          const __bf16* __restrict__ Bt, long bBatch,
                          const float* __restrict__ bias,
                          const float* __restrict__ aScale,
                          const float* __restrict__ aShift,
                          float* __restrict__ sum,
                          float* __restrict__ sumsq,
                          __bf16* __restrict__ Cout,
                          int M, int N, int K) {
    constexpr int LDSTR = 40;              // bf16 elements per LDS row (32 + pad)
    __shared__ __bf16 As[2][128 * LDSTR];
    __shared__ __bf16 Bs[2][128 * LDSTR];

    const int tid = threadIdx.x;
    const int e   = blockIdx.z;
    const int m0  = blockIdx.x * 128;
    const int n0  = blockIdx.y * 128;

    A    += (long)e * aBatch;
    Bt   += (long)e * bBatch;
    bias += (long)e * N;
    sum  += (long)e * N;
    sumsq+= (long)e * N;
    Cout += (long)e * M * N;
    const float* aS = aScale ? aScale + (long)e * K : nullptr;
    const float* aB = aShift ? aShift + (long)e * K : nullptr;

    const int wave = tid >> 5;
    const int lane = tid & 31;
    const int wr   = wave & 3;   // M group (0..3) -> rows 32*wr
    const int wc   = wave >> 2;  // N group (0..1) -> cols 64*wc
    const int ln16 = lane & 15;
    const int hf   = lane >> 4;  // 0 or 1

    v8f acc[2][4];
#pragma unroll
    for (int mi = 0; mi < 2; ++mi)
#pragma unroll
        for (int ni = 0; ni < 4; ++ni)
#pragma unroll
            for (int r = 0; r < 8; ++r) acc[mi][ni][r] = 0.f;

    // stage one 128x32 A tile + 128x32 B tile into LDS buffer `buf`
    auto stage = [&](int buf, int k0) {
#pragma unroll
        for (int it = 0; it < 2; ++it) {
            int c    = tid + it * 256;       // 512 chunks of 8 bf16
            int row  = c >> 2;
            int part = (c & 3) * 8;
            const __bf16* gA = A  + (long)(m0 + row) * K + k0 + part;
            const __bf16* gB = Bt + (long)(n0 + row) * K + k0 + part;
            __bf16* lA = &As[buf][row * LDSTR + part];
            __bf16* lB = &Bs[buf][row * LDSTR + part];
            if (aS) {
                // previous layer's BN+ReLU applied in registers while staging
                v8bf va = *(const v8bf*)gA;
#pragma unroll
                for (int i = 0; i < 8; ++i) {
                    int kk  = k0 + part + i;
                    float f = aS[kk] * (float)va[i] + aB[kk];
                    va[i]   = (__bf16)fmaxf(f, 0.f);
                }
                *(v8bf*)lA = va;
            } else {
                copy16_to_lds(gA, lA);       // async, no VGPR round-trip
            }
            copy16_to_lds(gB, lB);           // async, no VGPR round-trip
        }
    };

    const int nk = K / 32;
    stage(0, 0);
    async_wait_all();
    __syncthreads();

    for (int ks = 0; ks < nk; ++ks) {
        const int buf = ks & 1;
        if (ks + 1 < nk) stage(buf ^ 1, (ks + 1) * 32);   // prefetch next tile

        // ---- load fragments per ISA 16-bit A/B layouts ----
        v16bf af[2], bf_[4];
#pragma unroll
        for (int mi = 0; mi < 2; ++mi) {
            const __bf16* p = &As[buf][(wr * 32 + mi * 16 + ln16) * LDSTR + hf * 8];
            v8bf lo = *(const v8bf*)p;
            v8bf hi = *(const v8bf*)(p + 16);
#pragma unroll
            for (int i = 0; i < 8; ++i) { af[mi][i] = lo[i]; af[mi][i + 8] = hi[i]; }
        }
#pragma unroll
        for (int ni = 0; ni < 4; ++ni) {
            const __bf16* p = &Bs[buf][(wc * 64 + ni * 16 + ln16) * LDSTR + hf * 8];
            v8bf lo = *(const v8bf*)p;
            v8bf hi = *(const v8bf*)(p + 16);
#pragma unroll
            for (int i = 0; i < 8; ++i) { bf_[ni][i] = lo[i]; bf_[ni][i + 8] = hi[i]; }
        }

        // ---- 8 WMMAs per wave per k-step ----
#pragma unroll
        for (int mi = 0; mi < 2; ++mi)
#pragma unroll
            for (int ni = 0; ni < 4; ++ni)
                acc[mi][ni] = __builtin_amdgcn_wmma_f32_16x16x32_bf16(
                    false, af[mi], false, bf_[ni], (short)0, acc[mi][ni],
                    false, false);

        async_wait_all();     // my async copies for next tile are done
        __syncthreads();      // everyone's are; safe to flip buffers
    }

    // ---- epilogue: bias, store bf16 C, accumulate BN stats ----
    float sA[4], sQ[4];
#pragma unroll
    for (int ni = 0; ni < 4; ++ni) { sA[ni] = 0.f; sQ[ni] = 0.f; }
#pragma unroll
    for (int mi = 0; mi < 2; ++mi) {
#pragma unroll
        for (int ni = 0; ni < 4; ++ni) {
            int   n  = n0 + wc * 64 + ni * 16 + ln16;
            float bv = bias[n];
#pragma unroll
            for (int r = 0; r < 8; ++r) {
                int   m  = m0 + wr * 32 + mi * 16 + hf * 8 + r;
                float cv = acc[mi][ni][r] + bv;
                Cout[(long)m * N + n] = (__bf16)cv;
                sA[ni] += cv;
                sQ[ni] += cv * cv;
            }
        }
    }
#pragma unroll
    for (int ni = 0; ni < 4; ++ni) {
        float s = sA[ni] + __shfl_xor(sA[ni], 16, 32);
        float q = sQ[ni] + __shfl_xor(sQ[ni], 16, 32);
        if (hf == 0) {
            int n = n0 + wc * 64 + ni * 16 + ln16;
            atomicAdd(&sum[n], s);
            atomicAdd(&sumsq[n], q);
        }
    }
}

// ---------------------------------------------------------------------------
// Gates: logits[t][b][e] = x[b,:] . Wg[t,:,e] + bg[t,e]; softmax over e.
// One wave (32 lanes) per row b; E*T = 16 running sums per lane.
// ---------------------------------------------------------------------------
__global__ __launch_bounds__(256)
void gates_kernel(const float* __restrict__ x, const float* __restrict__ Wg,
                  const float* __restrict__ bg, float* __restrict__ gates,
                  int B, int D) {
    const int wave = threadIdx.x >> 5;
    const int lane = threadIdx.x & 31;
    const int b    = blockIdx.x * 8 + wave;

    float acc[16];
#pragma unroll
    for (int i = 0; i < 16; ++i) acc[i] = 0.f;

    for (int d = lane; d < D; d += 32) {
        float xv = x[(long)b * D + d];
        const float* w0 = Wg + (long)d * 8;            // [T][D][E], t=0
        const float* w1 = Wg + (long)(D + d) * 8;      // t=1
#pragma unroll
        for (int e2 = 0; e2 < 8; ++e2) {
            acc[e2]     += xv * w0[e2];
            acc[8 + e2] += xv * w1[e2];
        }
    }
#pragma unroll
    for (int off = 16; off >= 1; off >>= 1)
#pragma unroll
        for (int i = 0; i < 16; ++i) acc[i] += __shfl_xor(acc[i], off, 32);

    if (lane == 0) {
#pragma unroll
        for (int t = 0; t < 2; ++t) {
            float mx = -1e30f;
#pragma unroll
            for (int e2 = 0; e2 < 8; ++e2) {
                acc[t * 8 + e2] += bg[t * 8 + e2];
                mx = fmaxf(mx, acc[t * 8 + e2]);
            }
            float ex[8], s = 0.f;
#pragma unroll
            for (int e2 = 0; e2 < 8; ++e2) {
                ex[e2] = __expf(acc[t * 8 + e2] - mx);
                s += ex[e2];
            }
            float inv = 1.f / s;
#pragma unroll
            for (int e2 = 0; e2 < 8; ++e2)
                gates[((long)t * B + b) * 8 + e2] = ex[e2] * inv;
        }
    }
}

// ---------------------------------------------------------------------------
// Combine: out[t][b][e*H2+k] = relu(a2*h2[e][b][k]+b2) * gates[t][b][e]
// One thread per 4 consecutive k; writes both tasks with float4 stores.
// ---------------------------------------------------------------------------
__global__ __launch_bounds__(256)
void combine_kernel(const __bf16* __restrict__ h2,
                    const float* __restrict__ a2s, const float* __restrict__ b2s,
                    const float* __restrict__ gates, float* __restrict__ out,
                    int B) {
    long gid  = (long)blockIdx.x * 256 + threadIdx.x;
    long base = gid * 4;                 // flat index into [E][B][H2]
    int  k    = (int)(base & 255);       // H2 = 256
    long t2   = base >> 8;
    int  b    = (int)(t2 & (long)(B - 1)); // B = 32768 (pow2)
    int  e    = (int)(t2 >> 15);

    v4bf hv = *(const v4bf*)(h2 + base);
    float g0 = gates[(long)b * 8 + e];
    float g1 = gates[((long)B + b) * 8 + e];
    v4f f0, f1;
#pragma unroll
    for (int i = 0; i < 4; ++i) {
        float f = a2s[e * 256 + k + i] * (float)hv[i] + b2s[e * 256 + k + i];
        f = fmaxf(f, 0.f);
        f0[i] = f * g0;
        f1[i] = f * g1;
    }
    long co = (long)b * 2048 + (long)e * 256 + k;     // E*H2 = 2048
    *(v4f*)(out + co)                    = f0;
    *(v4f*)(out + (long)B * 2048 + co)   = f1;
}

// ---------------------------------------------------------------------------
// host-side launch
// ---------------------------------------------------------------------------
extern "C" void kernel_launch(void* const* d_in, const int* in_sizes, int n_in,
                              void* d_out, int out_size, void* d_ws, size_t ws_size,
                              hipStream_t stream) {
    (void)in_sizes; (void)n_in; (void)out_size; (void)ws_size;

    const float* x   = (const float*)d_in[0];
    const float* W1  = (const float*)d_in[1];
    const float* b1  = (const float*)d_in[2];
    const float* g1  = (const float*)d_in[3];
    const float* be1 = (const float*)d_in[4];
    const float* W2  = (const float*)d_in[5];
    const float* b2  = (const float*)d_in[6];
    const float* g2  = (const float*)d_in[7];
    const float* be2 = (const float*)d_in[8];
    const float* Wg  = (const float*)d_in[9];
    const float* bg  = (const float*)d_in[10];
    float* out = (float*)d_out;

    // workspace carving (256B aligned)
    char*  wsb = (char*)d_ws;
    size_t off = 0;
    auto carve = [&](size_t bytes) -> char* {
        char* p = wsb + off;
        off += (bytes + 255) & ~(size_t)255;
        return p;
    };
    __bf16* xb   = (__bf16*)carve((size_t)kB * kD * 2);
    __bf16* W1T  = (__bf16*)carve((size_t)kE * kH1 * kD * 2);
    __bf16* W2T  = (__bf16*)carve((size_t)kE * kH2 * kH1 * 2);
    __bf16* h1   = (__bf16*)carve((size_t)kE * kB * kH1 * 2);
    __bf16* h2   = (__bf16*)carve((size_t)kE * kB * kH2 * 2);
    float*  stats = (float*)carve(12288 * 4);   // sum1|sumsq1|sum2|sumsq2
    float*  sum1   = stats;
    float*  sumsq1 = stats + 4096;
    float*  sum2   = stats + 8192;
    float*  sumsq2 = stats + 10240;
    float*  a1s = (float*)carve(4096 * 4);
    float*  b1s = (float*)carve(4096 * 4);
    float*  a2s = (float*)carve(2048 * 4);
    float*  b2s = (float*)carve(2048 * 4);
    float*  gates = (float*)carve((size_t)kT * kB * kE * 4);

    // 0) zero BN stat accumulators
    zero_f32_kernel<<<48, 256, 0, stream>>>(stats, 12288);

    // 1) casts / transposes
    {
        long n4 = (long)kB * kD / 4;
        cast_f32_to_bf16_kernel<<<(unsigned)((n4 + 255) / 256), 256, 0, stream>>>(x, xb, n4);
        long t1 = (long)kE * kH1 * kD;
        transpose_cast_kernel<<<(unsigned)((t1 + 255) / 256), 256, 0, stream>>>(W1, W1T, kD, kH1, t1);
        long t2 = (long)kE * kH2 * kH1;
        transpose_cast_kernel<<<(unsigned)((t2 + 255) / 256), 256, 0, stream>>>(W2, W2T, kH1, kH2, t2);
    }

    // 2) expert layer 1: h1 = x @ W1 + b1  (+ BN1 stats)
    {
        dim3 grid(kB / 128, kH1 / 128, kE);
        gemm_bn_stats_kernel<<<grid, 256, 0, stream>>>(
            xb, 0L, W1T, (long)kH1 * kD, b1, nullptr, nullptr,
            sum1, sumsq1, h1, kB, kH1, kD);
    }
    bn_finalize_kernel<<<16, 256, 0, stream>>>(sum1, sumsq1, g1, be1, a1s, b1s,
                                               kE * kH1, 1.f / (float)kB);

    // 3) expert layer 2: h2 = relu(BN1(h1)) @ W2 + b2  (+ BN2 stats)
    {
        dim3 grid(kB / 128, kH2 / 128, kE);
        gemm_bn_stats_kernel<<<grid, 256, 0, stream>>>(
            h1, (long)kB * kH1, W2T, (long)kH2 * kH1, b2, a1s, b1s,
            sum2, sumsq2, h2, kB, kH2, kH1);
    }
    bn_finalize_kernel<<<8, 256, 0, stream>>>(sum2, sumsq2, g2, be2, a2s, b2s,
                                              kE * kH2, 1.f / (float)kB);

    // 4) gates (fp32, softmax over experts)
    gates_kernel<<<kB / 8, 256, 0, stream>>>(x, Wg, bg, gates, kB, kD);

    // 5) combine: BN2 + ReLU + gate weighting -> [T,B,E*H2]
    {
        long total4 = (long)kE * kB * kH2 / 4;
        combine_kernel<<<(unsigned)(total4 / 256), 256, 0, stream>>>(
            h2, a2s, b2s, gates, out, kB);
    }
}
